// SelfAttn2d_67121748902038
// MI455X (gfx1250) — matrix operand order repaired and checked by reading, not compiled, not run on hardware
//
#include <hip/hip_runtime.h>

// ---------------------------------------------------------------------------
// SelfAttn2d on MI455X (gfx1250): flash-attention with v_wmma_f32_16x16x32_f16
// + async Global->LDS staging (ASYNCcnt) for the V tiles.
// B=4, C=256, H=W=64 -> N=4096, DQK=32
// ---------------------------------------------------------------------------

#define B_ 4
#define C_ 256
#define N_ 4096
#define DQK_ 32
#define SCALE_INV 0.17677669529663687f /* 1/sqrt(32) */

typedef _Float16 half16 __attribute__((ext_vector_type(16)));
typedef float float8 __attribute__((ext_vector_type(8)));

union FragH {
  half16 v;
  _Float16 h[16];
  unsigned int u[8];
};

static __device__ inline float8 wmma_f16(const FragH& a, const FragH& b, float8 c) {
  return __builtin_amdgcn_wmma_f32_16x16x32_f16(false, a.v, false, b.v,
                                                (short)0, c, false, false);
}

// Async copy of 16 bytes Global -> LDS, per lane (CDNA5 GLOBAL_LOAD_ASYNC_TO_LDS).
static __device__ inline void async_copy_b128(unsigned lds_addr, const void* gptr) {
  asm volatile("global_load_async_to_lds_b128 %0, %1, off"
               :
               : "v"(lds_addr), "v"(gptr)
               : "memory");
}
static __device__ inline void wait_asynccnt0() {
  asm volatile("s_wait_asynccnt 0x0" ::: "memory");
}

// ---------------------------------------------------------------------------
// Kernel 1: top singular value via power iteration on A^T A (A staged in LDS).
// Block 0: wq [32x256], block 1: wk [32x256], block 2: wv [256x256].
// Writes invs[which] = 1/sigma_max.
// ---------------------------------------------------------------------------
__global__ __launch_bounds__(256) void sn_kernel(const float* __restrict__ wq,
                                                 const float* __restrict__ wk,
                                                 const float* __restrict__ wv,
                                                 float* __restrict__ invs) {
  extern __shared__ char smem[];
  float* Al = (float*)smem;          // up to 256 x 257 (padded rows)
  float* u = Al + 256 * 257;         // 256
  float* wvec = u + 256;             // 256
  float* red = wvec + 256;           // 256

  const int which = blockIdx.x;
  const float* A;
  int d;
  if (which == 0) { A = wq; d = 32; }
  else if (which == 1) { A = wk; d = 32; }
  else { A = wv; d = 256; }

  const int tid = threadIdx.x;
  for (int l = tid; l < d * 256; l += 256) {
    int r = l >> 8, c = l & 255;
    Al[r * 257 + c] = A[l];
  }
  u[tid] = 1.0f;
  __syncthreads();

  float lam = 1.0f;
  for (int it = 0; it < 512; ++it) {
    if (tid < d) {
      float s = 0.f;
      const float* ar = Al + tid * 257;
      for (int c = 0; c < 256; ++c) s += ar[c] * u[c];
      wvec[tid] = s;
    }
    __syncthreads();
    float up = 0.f;
    for (int j = 0; j < d; ++j) up += Al[j * 257 + tid] * wvec[j];
    red[tid] = up * up;
    __syncthreads();
    for (int s = 128; s > 0; s >>= 1) {
      if (tid < s) red[tid] += red[tid + s];
      __syncthreads();
    }
    float nrm = sqrtf(red[0]);  // ||A^T A u||; with u unit -> lambda = sigma^2
    lam = nrm;
    u[tid] = up / nrm;
    __syncthreads();
  }
  if (tid == 0) invs[which] = rsqrtf(lam);
}

// ---------------------------------------------------------------------------
// Kernel 2: fused projection  Out[j,n] = invsig * Wcat[j,:] . X[b,:,n]  (f16)
//   j in [0,32): Q, [32,64): K, [64,320): V.  One block = (b, 64-wide n tile),
//   8 waves x 10 subtiles of 16x16, K=256 in 8 WMMA steps.
// ---------------------------------------------------------------------------
__global__ __launch_bounds__(256) void proj_kernel(
    const float* __restrict__ x, const float* __restrict__ wq,
    const float* __restrict__ wk, const float* __restrict__ wv,
    const float* __restrict__ invs, _Float16* __restrict__ Qh,
    _Float16* __restrict__ Kh, _Float16* __restrict__ Vh) {
  const int b = blockIdx.x >> 6;
  const int nb = (blockIdx.x & 63) * 64;
  const int tid = threadIdx.x;
  const int lane = tid & 31;
  const int w = tid >> 5;
  const int lhi = (lane >= 16);
  const int l16 = lane & 15;

  const float* xb = x + (size_t)b * C_ * N_;

  for (int s = w * 10; s < w * 10 + 10; ++s) {
    const int jb = (s >> 2) * 16;  // 0..304
    const int nsub = s & 3;
    // per-lane A row (same for both lane halves)
    const int j = jb + l16;
    const float* wrow;
    float sc;
    if (j < 32) { wrow = wq + j * 256; sc = invs[0]; }
    else if (j < 64) { wrow = wk + (j - 32) * 256; sc = invs[1]; }
    else { wrow = wv + (j - 64) * 256; sc = invs[2]; }
    const int n = nb + nsub * 16 + l16;
    const float* xcol = xb + n;

    float8 dacc = {};
#pragma unroll
    for (int ks = 0; ks < 8; ++ks) {
      FragH af, bf;
      const int c0 = ks * 32;
      const int alo = lhi ? 8 : 0;
#pragma unroll
      for (int jj = 0; jj < 8; ++jj) {
        int coff = c0 + alo + ((jj < 4) ? 2 * jj : 8 + 2 * jj);
        af.h[2 * jj] = (_Float16)(wrow[coff] * sc);
        af.h[2 * jj + 1] = (_Float16)(wrow[coff + 1] * sc);
      }
      const int blo = c0 + (lhi ? 8 : 0);
#pragma unroll
      for (int h = 0; h < 16; ++h)
        bf.h[h] = (_Float16)xcol[(size_t)(blo + h + (h >= 8 ? 8 : 0)) * N_];  // coalesced over lanes
      dacc = wmma_f16(af, bf, dacc);
    }
    // D layout: VGPR r -> (M = r + 8*lhi, N = lane&15); M=j, N=n
#pragma unroll
    for (int r = 0; r < 8; ++r) {
      const int jj = jb + r + (lhi ? 8 : 0);
      const int nn = nb + nsub * 16 + l16;
      const _Float16 val = (_Float16)dacc[r];
      if (jj < 32) Qh[((size_t)b * N_ + nn) * DQK_ + jj] = val;
      else if (jj < 64) Kh[((size_t)b * N_ + nn) * DQK_ + (jj - 32)] = val;
      else Vh[((size_t)b * N_ + nn) * C_ + (jj - 64)] = val;
    }
  }
}

// ---------------------------------------------------------------------------
// Kernel 3: flash attention + residual.
//   Block = (b, 64-query tile). 8 wave32s. 64-key blocks, online softmax.
//   Wave w: S subtiles (si=w/2, sj={2(w&1),2(w&1)+1}); O columns [32w,32w+32).
//   V tile staged with async Global->LDS b128, overlapped with the S WMMAs.
// ---------------------------------------------------------------------------
__global__ __launch_bounds__(256) void attn_kernel(
    const _Float16* __restrict__ Qh, const _Float16* __restrict__ Kh,
    const _Float16* __restrict__ Vh, const float* __restrict__ x,
    const float* __restrict__ gamma, float* __restrict__ out) {
  extern __shared__ char smem[];
  float* S = (float*)smem;               // 64 x 65 (padded)
  float* rmax = S + 64 * 65;             // 64
  float* rsum = rmax + 64;               // 64
  float* ralpha = rsum + 64;             // 64
  _Float16* P = (_Float16*)(ralpha + 64);  // 64 x 72 (padded)
  _Float16* Vl = P + 64 * 72;              // 64 x 264 (padded), base 16B aligned

  const int b = blockIdx.x >> 6;
  const int mbase = (blockIdx.x & 63) * 64;
  const int tid = threadIdx.x;
  const int lane = tid & 31;
  const int w = tid >> 5;
  const int lhi = (lane >= 16);
  const int l16 = lane & 15;
  const int si = w >> 1;
  const int sj0 = (2 * w) & 3;

  if (tid < 64) { rmax[tid] = -1e30f; rsum[tid] = 0.f; }

  // Preload Q A-fragment for this wave's row group (rows 16*si).
  FragH qa;
  {
    const int m = mbase + 16 * si + l16;
    const _Float16* qrow = Qh + ((size_t)b * N_ + m) * DQK_;
    const int k0 = lhi ? 8 : 0;
#pragma unroll
    for (int j = 0; j < 8; ++j) {
      int koff = k0 + ((j < 4) ? 2 * j : 8 + 2 * j);
      qa.u[j] = *(const unsigned*)(qrow + koff);
    }
  }

  float8 acc[4][2];
#pragma unroll
  for (int ri = 0; ri < 4; ++ri)
#pragma unroll
    for (int ci = 0; ci < 2; ++ci) acc[ri][ci] = (float8){};

  for (int kb = 0; kb < N_; kb += 64) {
    __syncthreads();
    // ---- Issue async V staging: 64x256 f16 in 16B chunks, bypasses VGPRs.
    {
      const _Float16* vsrc = Vh + ((size_t)b * N_ + kb) * C_;
#pragma unroll
      for (int i = 0; i < 8; ++i) {
        const int ch = tid + i * 256;        // 2048 chunks of 8 halves
        const int key = ch >> 5;             // 32 chunks per key row
        const int eo = (ch & 31) << 3;       // half offset within row
        const unsigned lds = (unsigned)(size_t)(Vl + key * 264 + eo);
        async_copy_b128(lds, vsrc + key * C_ + eo);
      }
    }
    // ---- S = Q K^T / sqrt(d) while the async copies are in flight.
#pragma unroll
    for (int t2 = 0; t2 < 2; ++t2) {
      const int sj = sj0 + t2;
      FragH kf;
      const int key = kb + sj * 16 + l16;
      const _Float16* krow = Kh + ((size_t)b * N_ + key) * DQK_ + (lhi ? 8 : 0);
#pragma unroll
      for (int j = 0; j < 8; ++j) kf.u[j] = *(const unsigned*)(krow + ((j < 4) ? 2 * j : 8 + 2 * j));
      float8 sfr = {};
      sfr = wmma_f16(qa, kf, sfr);
#pragma unroll
      for (int r = 0; r < 8; ++r) {
        const int m = si * 16 + r + (lhi ? 8 : 0);
        S[m * 65 + sj * 16 + l16] = sfr[r] * SCALE_INV;
      }
    }
    wait_asynccnt0();  // our async V chunks are in LDS
    __syncthreads();
    // ---- Online softmax: 4 threads per row, quad-reduce via wave shuffles.
    {
      const int row = tid >> 2;
      const int seg = tid & 3;
      const float* srow = S + row * 65 + seg * 16;
      float pm = -1e30f;
#pragma unroll
      for (int j = 0; j < 16; ++j) pm = fmaxf(pm, srow[j]);
      pm = fmaxf(pm, __shfl_xor(pm, 1, 32));
      pm = fmaxf(pm, __shfl_xor(pm, 2, 32));
      const float mold = rmax[row];
      const float mnew = fmaxf(mold, pm);
      _Float16* prow = P + row * 72 + seg * 16;
      float ps = 0.f;
#pragma unroll
      for (int j = 0; j < 16; ++j) {
        const float p = __expf(srow[j] - mnew);
        ps += p;
        prow[j] = (_Float16)p;
      }
      ps += __shfl_xor(ps, 1, 32);
      ps += __shfl_xor(ps, 2, 32);
      if (seg == 0) {
        const float a = __expf(mold - mnew);
        rmax[row] = mnew;
        rsum[row] = rsum[row] * a + ps;
        ralpha[row] = a;
      }
    }
    __syncthreads();
    // ---- Rescale running O by alpha(row).
#pragma unroll
    for (int ri = 0; ri < 4; ++ri)
#pragma unroll
      for (int r = 0; r < 8; ++r) {
        const float a = ralpha[ri * 16 + r + (lhi ? 8 : 0)];
        acc[ri][0][r] *= a;
        acc[ri][1][r] *= a;
      }
    // ---- O += P (64x64) @ V (64x32 slice per wave): 16 WMMA per wave.
    FragH vf[2][2];  // [ci][kchunk]
#pragma unroll
    for (int ci = 0; ci < 2; ++ci) {
      const int e = w * 32 + ci * 16 + l16;
#pragma unroll
      for (int kc = 0; kc < 2; ++kc) {
        const int kb2 = kc * 32 + (lhi ? 16 : 0);
#pragma unroll
        for (int h = 0; h < 8; ++h) {
          unsigned lo = *(const unsigned short*)(Vl + (kb2 + 2 * h) * 264 + e);
          unsigned hi = *(const unsigned short*)(Vl + (kb2 + 2 * h + 1) * 264 + e);
          vf[ci][kc].u[h] = lo | (hi << 16);
        }
      }
    }
#pragma unroll
    for (int ri = 0; ri < 4; ++ri) {
      FragH pa[2];
      const _Float16* prow = P + (ri * 16 + l16) * 72 + (lhi ? 16 : 0);
#pragma unroll
      for (int kc = 0; kc < 2; ++kc)
#pragma unroll
        for (int j = 0; j < 8; ++j) {
          int koff = kc * 32 + 2 * j;
          pa[kc].u[j] = *(const unsigned*)(prow + koff);
        }
#pragma unroll
      for (int ci = 0; ci < 2; ++ci) {
        acc[ri][ci] = wmma_f16(pa[0], vf[ci][0], acc[ri][ci]);
        acc[ri][ci] = wmma_f16(pa[1], vf[ci][1], acc[ri][ci]);
      }
    }
  }

  // Normalize by running sum.
  __syncthreads();
  if (tid < 64) ralpha[tid] = 1.0f / rsum[tid];
  __syncthreads();
#pragma unroll
  for (int ri = 0; ri < 4; ++ri)
#pragma unroll
    for (int r = 0; r < 8; ++r) {
      const float a = ralpha[ri * 16 + r + (lhi ? 8 : 0)];
      acc[ri][0][r] *= a;
      acc[ri][1][r] *= a;
    }

  // Epilogue: out[b,e,m] = x[b,e,m] + gamma * O[m,e], transposed via LDS for
  // coalesced global stores. 4 groups of 64 e-columns.
  const float g = gamma[0];
  for (int gi = 0; gi < 4; ++gi) {
    if (si == gi) {
#pragma unroll
      for (int ri = 0; ri < 4; ++ri)
#pragma unroll
        for (int ci = 0; ci < 2; ++ci)
#pragma unroll
          for (int r = 0; r < 8; ++r) {
            const int m = ri * 16 + r + (lhi ? 8 : 0);
            const int el = (w & 1) * 32 + ci * 16 + l16;
            S[m * 65 + el] = acc[ri][ci][r];
          }
    }
    __syncthreads();
    {
      const int el = tid >> 2;
      const int m0 = (tid & 3) * 16;
      const size_t base = ((size_t)b * C_ + gi * 64 + el) * N_ + mbase + m0;
      const float* xr = x + base;
      float* orow = out + base;
#pragma unroll
      for (int i = 0; i < 16; ++i) orow[i] = xr[i] + g * S[(m0 + i) * 65 + el];
    }
    __syncthreads();
  }
}

// ---------------------------------------------------------------------------
extern "C" void kernel_launch(void* const* d_in, const int* in_sizes, int n_in,
                              void* d_out, int out_size, void* d_ws,
                              size_t ws_size, hipStream_t stream) {
  const float* x = (const float*)d_in[0];
  const float* wq = (const float*)d_in[1];
  const float* wk = (const float*)d_in[2];
  const float* wv = (const float*)d_in[3];
  const float* gamma = (const float*)d_in[4];
  float* out = (float*)d_out;

  char* ws = (char*)d_ws;
  float* invs = (float*)ws;  // 3 floats (padded to 256B)
  _Float16* Qh = (_Float16*)(ws + 256);
  _Float16* Kh = Qh + (size_t)B_ * N_ * DQK_;
  _Float16* Vh = Kh + (size_t)B_ * N_ * DQK_;
  // total ws use: 256 + 2*1MB + 8MB ≈ 10.5 MB

  const size_t sn_lds = (size_t)(256 * 257 + 768) * 4;  // ~260 KB (<320 KB WGP LDS)
  sn_kernel<<<3, 256, sn_lds, stream>>>(wq, wk, wv, invs);

  proj_kernel<<<B_ * (N_ / 64), 256, 0, stream>>>(x, wq, wk, wv, invs, Qh, Kh, Vh);

  const size_t at_lds =
      (size_t)(64 * 65 + 192) * 4 + (size_t)(64 * 72 + 64 * 264) * 2;  // 60416 B
  attn_kernel<<<B_ * (N_ / 64), 256, at_lds, stream>>>(Qh, Kh, Vh, x, gamma, out);
}
